// HybridNoAttention_64166811402355
// MI455X (gfx1250) — compile-verified, matrix operand order checked by reading, not run on hardware
//
#include <hip/hip_runtime.h>
#include <hip/hip_bf16.h>
#include <math.h>

typedef __bf16 bf16_t;
typedef __attribute__((ext_vector_type(16))) __bf16 v16bf;
typedef __attribute__((ext_vector_type(8)))  __bf16 v8bf;
typedef __attribute__((ext_vector_type(8)))  float  v8f;

union FragBF { v16bf v; v8bf h[2]; };
union AccF   { v8f  v; float f[8]; };

// CDNA5 16-bit A/B fragment load (16x32 / 32x16), row-major source, ld in elements.
// lane<16 holds K {kb..kb+7, 16+kb..16+kb+7} with kb=0; lane>=16 with kb=8.
__device__ __forceinline__ v16bf load_frag(const bf16_t* base, int ld, int lane) {
  const int row = lane & 15;
  const int kb  = (lane >> 4) << 3;
  const bf16_t* p = base + (size_t)row * (size_t)ld + kb;
  FragBF u;
  u.h[0] = *(const v8bf*)(p);
  u.h[1] = *(const v8bf*)(p + 16);
  return u.v;
}

// ---------------------------------------------------------------- conversions
__global__ void pad_bf16_kernel(const float* __restrict__ src, bf16_t* __restrict__ dst,
                                int rows, int k, int kp) {
  size_t e = (size_t)blockIdx.x * blockDim.x + threadIdx.x;
  size_t tot = (size_t)rows * kp;
  if (e >= tot) return;
  int r = (int)(e / kp);
  int j = (int)(e % kp);
  dst[e] = (j < k) ? (bf16_t)src[(size_t)r * k + j] : (bf16_t)0.f;
}

// ---------------------------------------------------------------- quantile 0.8
__global__ void quantile_kernel(const float* __restrict__ fc, float* __restrict__ thr, int N) {
  const int b = blockIdx.x;
  const float* m = fc + (size_t)b * N * N;
  const int n2 = N * N;
  __shared__ float red[256];
  __shared__ unsigned hist[1024];
  const int tid = threadIdx.x;
  float mx = 0.f;
  for (int i = tid; i < n2; i += 256) mx = fmaxf(mx, fabsf(m[i]));
  red[tid] = mx; __syncthreads();
  for (int s = 128; s > 0; s >>= 1) { if (tid < s) red[tid] = fmaxf(red[tid], red[tid + s]); __syncthreads(); }
  const float gmx = red[0] * 1.0000001f + 1e-12f;
  for (int i = tid; i < 1024; i += 256) hist[i] = 0u;
  __syncthreads();
  for (int i = tid; i < n2; i += 256) {
    float a = fabsf(m[i]);
    int bin = (int)(a * (1024.f / gmx));
    if (bin > 1023) bin = 1023;
    atomicAdd(&hist[bin], 1u);
  }
  __syncthreads();
  if (tid == 0) {
    const int target = (int)(0.8f * (float)n2);
    unsigned cum = 0; int bsel = 1023;
    for (int i = 0; i < 1024; ++i) { cum += hist[i]; if ((int)cum >= target) { bsel = i; break; } }
    thr[b] = ((float)bsel + 0.5f) * gmx * (1.f / 1024.f);
  }
}

// ------------------------------------------------- per-row stats: rowsum(A), degree, local_eff
__global__ void rowstats_kernel(const float* __restrict__ fc, const float* __restrict__ thr,
                                float* __restrict__ rowsumA, float* __restrict__ degree,
                                float* __restrict__ loceff, int N) {
  const int idx = blockIdx.x;
  const int b = idx / N, i = idx % N;
  const float* row = fc + ((size_t)b * N + i) * N;
  const float t = thr[b];
  __shared__ float s1[64], s2[64];
  const int tid = threadIdx.x;
  float sa = 0.f, sd = 0.f;
  for (int j = tid; j < N; j += 64) {
    float a = fabsf(row[j]);
    sd += a;
    sa += (a > t) ? 1.f : 0.f;
    if (j == i) sa += 1.f;  // + eye
  }
  s1[tid] = sa; s2[tid] = sd; __syncthreads();
  for (int s = 32; s > 0; s >>= 1) { if (tid < s) { s1[tid] += s1[tid + s]; s2[tid] += s2[tid + s]; } __syncthreads(); }
  if (tid == 0) { rowsumA[idx] = s1[0]; degree[idx] = s2[0]; loceff[idx] = s2[0] / (float)N; }
}

// ------------------------------------------------------------- normalized adjacency
__global__ void anorm_kernel(const float* __restrict__ fc, const float* __restrict__ thr,
                             const float* __restrict__ rowsumA, float* __restrict__ An, int N) {
  const int idx = blockIdx.x;
  const int b = idx / N, i = idx % N;
  const float* row = fc + ((size_t)b * N + i) * N;
  const float t = thr[b];
  const float di = rsqrtf(rowsumA[idx]);
  const float* rs = rowsumA + (size_t)b * N;
  float* out = An + ((size_t)b * N + i) * N;
  for (int j = threadIdx.x; j < N; j += 256) {
    float a = (fabsf(row[j]) > t) ? 1.f : 0.f;
    if (j == i) a += 1.f;
    out[j] = a * di * rsqrtf(rs[j]);
  }
}

// ------------------------------------------------------------- clustering coefficient
// clus[b,i] = sum_j fc[i,j] * dot(fc_row_j, fc_row_i) / (deg + 1e-8)   (fc symmetric)
__global__ void clustering_kernel(const float* __restrict__ fc, const float* __restrict__ degree,
                                  float* __restrict__ clus, int N) {
  const int idx = blockIdx.x;
  const int b = idx / N, i = idx % N;
  const float* M = fc + (size_t)b * N * N;
  __shared__ float rowi[256];
  __shared__ float red[256];
  const int tid = threadIdx.x;
  for (int k = tid; k < N; k += 256) rowi[k] = M[(size_t)i * N + k];
  __syncthreads();
  float acc = 0.f;
  for (int j = tid; j < N; j += 256) {
    const float* rowj = M + (size_t)j * N;
    float d = 0.f;
    for (int k = 0; k < N; ++k) d += rowj[k] * rowi[k];
    acc += rowi[j] * d;
  }
  red[tid] = acc; __syncthreads();
  for (int s = 128; s > 0; s >>= 1) { if (tid < s) red[tid] += red[tid + s]; __syncthreads(); }
  if (tid == 0) clus[idx] = red[0] / (degree[idx] + 1e-8f);
}

// ------------------------------------------------------- dominant eigenvector (shifted power iter)
__global__ void power_kernel(const float* __restrict__ fc, const float* __restrict__ degree,
                             float* __restrict__ eig, int N, int iters) {
  const int b = blockIdx.x;
  const float* M = fc + (size_t)b * N * N;
  __shared__ float v[256], wv[256], red[256];
  const int tid = threadIdx.x;
  float mx = 0.f;
  for (int i = tid; i < N; i += 256) mx = fmaxf(mx, degree[b * N + i]);
  red[tid] = mx; __syncthreads();
  for (int s = 128; s > 0; s >>= 1) { if (tid < s) red[tid] = fmaxf(red[tid], red[tid + s]); __syncthreads(); }
  const float sigma = red[0] + 1.f;   // Gershgorin shift -> largest algebraic eig dominates
  v[tid] = (tid < N) ? rsqrtf((float)N) : 0.f;
  __syncthreads();
  for (int it = 0; it < iters; ++it) {
    for (int i = tid; i < N; i += 256) {
      const float* row = M + (size_t)i * N;
      float s = sigma * v[i];
      for (int k = 0; k < N; ++k) s += row[k] * v[k];
      wv[i] = s;
    }
    __syncthreads();
    float loc = 0.f;
    for (int i = tid; i < N; i += 256) loc += wv[i] * wv[i];
    red[tid] = loc; __syncthreads();
    for (int s = 128; s > 0; s >>= 1) { if (tid < s) red[tid] += red[tid + s]; __syncthreads(); }
    const float inv = rsqrtf(red[0] + 1e-20f);
    __syncthreads();
    for (int i = tid; i < N; i += 256) v[i] = wv[i] * inv;
    __syncthreads();
  }
  for (int i = tid; i < N; i += 256) eig[b * N + i] = fabsf(v[i]);
}

// ------------------------------------------------------------- node feature assembly
__global__ void feat_kernel(const float* __restrict__ deg, const float* __restrict__ clus,
                            const float* __restrict__ eg, const float* __restrict__ loc,
                            float* __restrict__ h0, int total) {
  int e = blockIdx.x * blockDim.x + threadIdx.x;
  if (e >= total) return;
  h0[4 * e + 0] = deg[e];
  h0[4 * e + 1] = clus[e];
  h0[4 * e + 2] = eg[e];
  h0[4 * e + 3] = loc[e];
}

// ------------------------------------------------------------- GNN: xw = h @ W
__global__ void xw_kernel(const float* __restrict__ h, const float* __restrict__ W,
                          float* __restrict__ out, int in_dim, int out_dim) {
  const int idx = blockIdx.x;
  const int f = threadIdx.x;
  const float* hv = h + (size_t)idx * in_dim;
  float s = 0.f;
  for (int c = 0; c < in_dim; ++c) s += hv[c] * W[(size_t)c * out_dim + f];
  out[(size_t)idx * out_dim + f] = s;
}

// ------------------------------------------------------------- GNN: relu(A_norm @ xw + b)
__global__ void agg_kernel(const float* __restrict__ An, const float* __restrict__ xw,
                           const float* __restrict__ bias, float* __restrict__ out,
                           int N, int out_dim) {
  const int idx = blockIdx.x;
  const int b = idx / N, i = idx % N;
  const int f = threadIdx.x;
  const float* arow = An + ((size_t)b * N + i) * N;
  const float* xb = xw + (size_t)b * N * out_dim;
  float s = bias[f];
  for (int j = 0; j < N; ++j) s += arow[j] * xb[(size_t)j * out_dim + f];
  out[(size_t)idx * out_dim + f] = fmaxf(s, 0.f);
}

__global__ void pool_kernel(const float* __restrict__ h, float* __restrict__ emb, int N, int F) {
  const int b = blockIdx.x, f = threadIdx.x;
  float s = 0.f;
  for (int i = 0; i < N; ++i) s += h[((size_t)b * N + i) * F + f];
  emb[b * F + f] = s / (float)N;
}

// ------------------------------------------------------------- persistent bf16-WMMA LSTM
// One workgroup (32 waves) owns the whole B(128) x 4H(512) gate GEMM each timestep:
//   gates = (b_ih+b_hh) + x_t @ Wih^T + h_{t-1} @ Whh^T    via v_wmma_f32_16x16x32_bf16
// Each wave owns one 16-wide gate column (n0) and walks 8 batch M-tiles in two
// groups of 4: one B fragment is loaded per k-step and reused by 4 accumulators,
// and all 4 A fragments are issued before the 4 wmmas so the waits stagger.
__global__ __launch_bounds__(1024)
void lstm_kernel(const bf16_t* __restrict__ x, int Kx,            // [B][T][Kx] bf16, Kx % 32 == 0
                 const bf16_t* __restrict__ Wih,                  // [4H][Kx] bf16
                 const bf16_t* __restrict__ Whh,                  // [4H][H]  bf16
                 const float* __restrict__ bih, const float* __restrict__ bhh,
                 float* __restrict__ cbuf, float* __restrict__ gbuf,  // [B][H], [B][4H]
                 int T, int rev,
                 bf16_t* __restrict__ out_seq, int out_ld, int out_col,
                 float* __restrict__ out_fin, int fin_ld, int fin_col) {
  const int H = 128, NB = 128, G = 512;
  __shared__ bf16_t hs[NB * H];                                   // 32 KB: h_{t-1} in bf16
  const int tid  = threadIdx.x;
  const int lane = tid & 31;
  const int wave = tid >> 5;

  for (int e = tid; e < NB * H; e += 1024) { hs[e] = (bf16_t)0.f; cbuf[e] = 0.f; }
  __syncthreads();

  const int n0 = wave * 16;        // 32 waves <-> 32 gate tiles
  const int cn = lane & 15;
  const float bias = bih[n0 + cn] + bhh[n0 + cn];
  const int lda = T * Kx;

  for (int ts = 0; ts < T; ++ts) {
    const int t = rev ? (T - 1 - ts) : ts;

    // prefetch next timestep's activations (global_prefetch_b8), 1 cacheline/thread
    if (ts + 1 < T) {
      const int tn = rev ? (t - 1) : (t + 1);
      const int prow = tid >> 3;          // 128 batch rows over 1024 threads
      const int poff = (tid & 7) << 6;    // 64B granules across the Kx*2-byte row
      if (poff < Kx * 2)
        __builtin_prefetch((const char*)(x + ((size_t)prow * T + tn) * Kx) + poff, 0, 0);
    }

    // --- GEMM phase
    for (int mg = 0; mg < 2; ++mg) {      // two groups of 4 M-tiles
      AccF acc[4];
      #pragma unroll
      for (int mt = 0; mt < 4; ++mt)
        #pragma unroll
        for (int vv = 0; vv < 8; ++vv) acc[mt].f[vv] = bias;

      {   // x_t @ Wih^T : issue B + all 4 A fragment loads, then drain into 4 wmmas
        const bf16_t* bbase = Wih + (size_t)n0 * Kx;
        const bf16_t* abase = x + ((size_t)(mg * 64) * T + t) * Kx;
        for (int k = 0; k < Kx; k += 32) {
          v16bf bfrag = load_frag(bbase + k, Kx, lane);
          v16bf afr[4];
          #pragma unroll
          for (int mt = 0; mt < 4; ++mt)
            afr[mt] = load_frag(abase + (size_t)(mt * 16) * lda + k, lda, lane);
          #pragma unroll
          for (int mt = 0; mt < 4; ++mt)
            acc[mt].v = __builtin_amdgcn_wmma_f32_16x16x32_bf16(false, afr[mt], false, bfrag,
                                                                (short)0, acc[mt].v, false, false);
        }
      }
      {   // h_{t-1} @ Whh^T (h fragments from LDS), same issue-then-drain shape
        const bf16_t* bbase = Whh + (size_t)n0 * H;
        const bf16_t* abase = hs + mg * 64 * H;
        for (int k = 0; k < H; k += 32) {
          v16bf bfrag = load_frag(bbase + k, H, lane);
          v16bf afr[4];
          #pragma unroll
          for (int mt = 0; mt < 4; ++mt)
            afr[mt] = load_frag(abase + mt * 16 * H + k, H, lane);
          #pragma unroll
          for (int mt = 0; mt < 4; ++mt)
            acc[mt].v = __builtin_amdgcn_wmma_f32_16x16x32_bf16(false, afr[mt], false, bfrag,
                                                                (short)0, acc[mt].v, false, false);
        }
      }
      const int mbl = (lane >> 4) << 3;
      #pragma unroll
      for (int mt = 0; mt < 4; ++mt) {
        const int mb = (mg * 4 + mt) * 16 + mbl;
        #pragma unroll
        for (int vv = 0; vv < 8; ++vv)
          gbuf[(size_t)(mb + vv) * G + n0 + cn] = acc[mt].f[vv];
      }
    }
    __syncthreads();

    // --- cell phase
    for (int e = tid; e < NB * H; e += 1024) {
      const int b  = e >> 7;
      const int hh = e & 127;
      const float* g = gbuf + (size_t)b * G;
      const float gi = g[hh], gf = g[H + hh], gg = g[2 * H + hh], go = g[3 * H + hh];
      float c = cbuf[e];
      const float si = 1.f / (1.f + __expf(-gi));
      const float sf = 1.f / (1.f + __expf(-gf));
      const float so = 1.f / (1.f + __expf(-go));
      c = sf * c + si * tanhf(gg);
      const float hv = so * tanhf(c);
      cbuf[e] = c;
      hs[e] = (bf16_t)hv;
      if (out_seq) out_seq[((size_t)b * T + t) * out_ld + out_col + hh] = (bf16_t)hv;
      if (out_fin && ts == T - 1) out_fin[(size_t)b * fin_ld + fin_col + hh] = hv;
    }
    __syncthreads();
  }
}

// ------------------------------------------------------------- fusion + classifier head
__global__ void head_kernel(const float* __restrict__ emb, const float* __restrict__ stan,
                            const float* __restrict__ fw, const float* __restrict__ fb,
                            const float* __restrict__ k1w, const float* __restrict__ k1b,
                            const float* __restrict__ gam, const float* __restrict__ bet,
                            const float* __restrict__ k2w, const float* __restrict__ k2b,
                            float* __restrict__ out) {
  const int b = blockIdx.x;
  const int tid = threadIdx.x;
  __shared__ float feat[288], fused[128], z[64];
  for (int k = tid; k < 288; k += 128)
    feat[k] = (k < 32) ? emb[b * 32 + k] : stan[b * 256 + (k - 32)];
  __syncthreads();
  float s = fb[tid];
  for (int k = 0; k < 288; ++k) s += feat[k] * fw[k * 128 + tid];
  fused[tid] = fmaxf(s, 0.f);
  __syncthreads();
  if (tid < 64) {
    float s2 = k1b[tid];
    for (int k = 0; k < 128; ++k) s2 += fused[k] * k1w[k * 64 + tid];
    s2 = fmaxf(s2, 0.f);
    z[tid] = s2 * (gam[tid] * rsqrtf(1.f + 1e-5f)) + bet[tid];
  }
  __syncthreads();
  if (tid < 2) {
    float s3 = k2b[tid];
    for (int k = 0; k < 64; ++k) s3 += z[k] * k2w[k * 2 + tid];
    out[b * 2 + tid] = s3;
  }
}

// ============================================================================
extern "C" void kernel_launch(void* const* d_in, const int* in_sizes, int n_in,
                              void* d_out, int out_size, void* d_ws, size_t ws_size,
                              hipStream_t stream) {
  (void)in_sizes; (void)n_in; (void)out_size; (void)ws_size;
  const int B = 128, T = 512, N = 200, H = 128;
  const int KX0 = 224;   // 200 zero-padded to 7*32
  const int KX1 = 256;   // 2H, already 8*32

  const float* fc    = (const float*)d_in[0];
  const float* roi   = (const float*)d_in[1];
  const float* gw1   = (const float*)d_in[2];  const float* gb1 = (const float*)d_in[3];
  const float* gw2   = (const float*)d_in[4];  const float* gb2 = (const float*)d_in[5];
  const float* gw3   = (const float*)d_in[6];  const float* gb3 = (const float*)d_in[7];
  const float* wih0f = (const float*)d_in[8];  const float* whh0f = (const float*)d_in[9];
  const float* bih0f = (const float*)d_in[10]; const float* bhh0f = (const float*)d_in[11];
  const float* wih0b = (const float*)d_in[12]; const float* whh0b = (const float*)d_in[13];
  const float* bih0b = (const float*)d_in[14]; const float* bhh0b = (const float*)d_in[15];
  const float* wih1f = (const float*)d_in[16]; const float* whh1f = (const float*)d_in[17];
  const float* bih1f = (const float*)d_in[18]; const float* bhh1f = (const float*)d_in[19];
  const float* wih1b = (const float*)d_in[20]; const float* whh1b = (const float*)d_in[21];
  const float* bih1b = (const float*)d_in[22]; const float* bhh1b = (const float*)d_in[23];
  const float* fw    = (const float*)d_in[24]; const float* fbv = (const float*)d_in[25];
  const float* k1w   = (const float*)d_in[26]; const float* k1b = (const float*)d_in[27];
  const float* gam   = (const float*)d_in[28]; const float* bet = (const float*)d_in[29];
  const float* k2w   = (const float*)d_in[30]; const float* k2b = (const float*)d_in[31];
  float* out = (float*)d_out;

  // ---- workspace carve-out (all 256B aligned)
  char* wsb = (char*)d_ws;
  size_t off = 0;
  auto alloc = [&](size_t bytes) -> char* {
    char* p = wsb + off;
    off = (off + bytes + 255) & ~(size_t)255;
    return p;
  };
  bf16_t* xb      = (bf16_t*)alloc((size_t)B * T * KX0 * sizeof(bf16_t));   // 29.4 MB
  bf16_t* x1b     = (bf16_t*)alloc((size_t)B * T * KX1 * sizeof(bf16_t));   // 33.6 MB
  bf16_t* wih0f_b = (bf16_t*)alloc((size_t)512 * KX0 * sizeof(bf16_t));
  bf16_t* wih0b_b = (bf16_t*)alloc((size_t)512 * KX0 * sizeof(bf16_t));
  bf16_t* whh0f_b = (bf16_t*)alloc((size_t)512 * H * sizeof(bf16_t));
  bf16_t* whh0b_b = (bf16_t*)alloc((size_t)512 * H * sizeof(bf16_t));
  bf16_t* wih1f_b = (bf16_t*)alloc((size_t)512 * KX1 * sizeof(bf16_t));
  bf16_t* wih1b_b = (bf16_t*)alloc((size_t)512 * KX1 * sizeof(bf16_t));
  bf16_t* whh1f_b = (bf16_t*)alloc((size_t)512 * H * sizeof(bf16_t));
  bf16_t* whh1b_b = (bf16_t*)alloc((size_t)512 * H * sizeof(bf16_t));
  float* An      = (float*)alloc((size_t)B * N * N * sizeof(float));        // 20.5 MB
  float* thr     = (float*)alloc((size_t)B * sizeof(float));
  float* rowsumA = (float*)alloc((size_t)B * N * sizeof(float));
  float* degree  = (float*)alloc((size_t)B * N * sizeof(float));
  float* clus    = (float*)alloc((size_t)B * N * sizeof(float));
  float* eigc    = (float*)alloc((size_t)B * N * sizeof(float));
  float* loceff  = (float*)alloc((size_t)B * N * sizeof(float));
  float* h0feat  = (float*)alloc((size_t)B * N * 4 * sizeof(float));
  float* gbufA   = (float*)alloc((size_t)B * N * 128 * sizeof(float));      // 13.1 MB act
  float* gbufB   = (float*)alloc((size_t)B * N * 128 * sizeof(float));      // 13.1 MB tmp
  float* emb     = (float*)alloc((size_t)B * 32 * sizeof(float));
  float* stan    = (float*)alloc((size_t)B * 256 * sizeof(float));
  float* cbuf    = (float*)alloc((size_t)B * H * sizeof(float));
  float* gates   = (float*)alloc((size_t)B * 512 * sizeof(float));

  auto pad = [&](const float* s, bf16_t* d, int rows, int k, int kp) {
    size_t tot = (size_t)rows * kp;
    int blocks = (int)((tot + 255) / 256);
    pad_bf16_kernel<<<blocks, 256, 0, stream>>>(s, d, rows, k, kp);
  };

  // ---- bf16 conversions (zero-padded K)
  pad(roi,   xb,      B * T, N,   KX0);
  pad(wih0f, wih0f_b, 512,   N,   KX0);
  pad(wih0b, wih0b_b, 512,   N,   KX0);
  pad(whh0f, whh0f_b, 512,   H,   H);
  pad(whh0b, whh0b_b, 512,   H,   H);
  pad(wih1f, wih1f_b, 512,   KX1, KX1);
  pad(wih1b, wih1b_b, 512,   KX1, KX1);
  pad(whh1f, whh1f_b, 512,   H,   H);
  pad(whh1b, whh1b_b, 512,   H,   H);

  // ---- graph branch
  quantile_kernel<<<B, 256, 0, stream>>>(fc, thr, N);
  rowstats_kernel<<<B * N, 64, 0, stream>>>(fc, thr, rowsumA, degree, loceff, N);
  anorm_kernel<<<B * N, 256, 0, stream>>>(fc, thr, rowsumA, An, N);
  clustering_kernel<<<B * N, 256, 0, stream>>>(fc, degree, clus, N);
  power_kernel<<<B, 256, 0, stream>>>(fc, degree, eigc, N, 64);
  feat_kernel<<<(B * N + 255) / 256, 256, 0, stream>>>(degree, clus, eigc, loceff, h0feat, B * N);

  xw_kernel<<<B * N, 128, 0, stream>>>(h0feat, gw1, gbufB, 4, 128);
  agg_kernel<<<B * N, 128, 0, stream>>>(An, gbufB, gb1, gbufA, N, 128);
  xw_kernel<<<B * N, 64, 0, stream>>>(gbufA, gw2, gbufB, 128, 64);
  agg_kernel<<<B * N, 64, 0, stream>>>(An, gbufB, gb2, gbufA, N, 64);
  xw_kernel<<<B * N, 32, 0, stream>>>(gbufA, gw3, gbufB, 64, 32);
  agg_kernel<<<B * N, 32, 0, stream>>>(An, gbufB, gb3, gbufA, N, 32);
  pool_kernel<<<B, 32, 0, stream>>>(gbufA, emb, N, 32);

  // ---- BiLSTM stack (persistent WMMA kernels; layer0 writes bf16 x1 directly)
  lstm_kernel<<<1, 1024, 0, stream>>>(xb, KX0, wih0f_b, whh0f_b, bih0f, bhh0f,
                                      cbuf, gates, T, 0,
                                      x1b, KX1, 0, (float*)nullptr, 0, 0);
  lstm_kernel<<<1, 1024, 0, stream>>>(xb, KX0, wih0b_b, whh0b_b, bih0b, bhh0b,
                                      cbuf, gates, T, 1,
                                      x1b, KX1, 128, (float*)nullptr, 0, 0);
  lstm_kernel<<<1, 1024, 0, stream>>>(x1b, KX1, wih1f_b, whh1f_b, bih1f, bhh1f,
                                      cbuf, gates, T, 0,
                                      (bf16_t*)nullptr, 0, 0, stan, 256, 0);
  lstm_kernel<<<1, 1024, 0, stream>>>(x1b, KX1, wih1b_b, whh1b_b, bih1b, bhh1b,
                                      cbuf, gates, T, 1,
                                      (bf16_t*)nullptr, 0, 0, stan, 256, 128);

  // ---- fusion + classifier
  head_kernel<<<B, 128, 0, stream>>>(emb, stan, fw, fbv, k1w, k1b, gam, bet, k2w, k2b, out);
}
